// SimpleFusion_26259430048535
// MI455X (gfx1250) — matrix operand order repaired and checked by reading, not compiled, not run on hardware
//
#include <hip/hip_runtime.h>

// ---------------------------------------------------------------------------
// PointFusion: bilinear BEV gather + concat + Linear(432->128) + BN + ReLU
// GEMM done with CDNA5 v_wmma_f32_16x16x32_bf16 (f32 accumulate).
// Round 2: software-pipelined LDS fragment loads (4-deep B buffer, 2-deep A)
//          + async global->LDS staging of the bf16 weight tile (ASYNCcnt).
// ---------------------------------------------------------------------------

#define N_PTS      262144
#define BATCH      4
#define C_BEV      256
#define H_G        200
#define W_G        176
#define PLANE      (H_G * W_G)          // 35200
#define C_PT       176                  // 16+32+64+64
#define C_IN       432
#define K_PAD      448                  // 14 * 32
#define KT_N       14                   // K tiles
#define C_OUT      128
#define ROW_LDS    456                  // padded row (halfwords): 16B aligned, conflict-free
#define TILE_M     128                  // points per workgroup
#define BN_EPS     1e-5f

typedef __bf16       bf16x16 __attribute__((ext_vector_type(16)));
typedef float        f32x8   __attribute__((ext_vector_type(8)));
typedef float        f32x4   __attribute__((ext_vector_type(4)));
typedef unsigned int u32x4   __attribute__((ext_vector_type(4)));
typedef unsigned int u32x2   __attribute__((ext_vector_type(2)));

union BF16Frag {
    struct { u32x4 lo; u32x4 hi; } u;
    bf16x16 v;
};

static __device__ __forceinline__ unsigned short f32_to_bf16(float f) {
    unsigned int u = __float_as_uint(f);
    u = (u + 0x7FFFu + ((u >> 16) & 1u)) >> 16;   // round-to-nearest-even
    return (unsigned short)u;
}

// ---------------------------------------------------------------------------
// Kernel 0: W [128,432] f32 -> bf16, K padded with zeros to 448.
// ---------------------------------------------------------------------------
__global__ __launch_bounds__(256) void wcvt_kernel(const float* __restrict__ W,
                                                   unsigned short* __restrict__ Wbf) {
    int i = blockIdx.x * 256 + threadIdx.x;       // 128*448 = 57344 elements
    int o = i / K_PAD;
    int k = i - o * K_PAD;
    unsigned short v = 0;
    if (k < C_IN) v = f32_to_bf16(W[o * C_IN + k]);
    Wbf[i] = v;
}

// ---------------------------------------------------------------------------
// Kernel 1: fused gather + GEMM + BN + ReLU.
// 256 threads = 8 waves (wave32). Each WG: 128 points, all 128 out channels.
// ---------------------------------------------------------------------------
__global__ __launch_bounds__(256) void fusion_kernel(
    const float* __restrict__ coords,      // [N,4]
    const float* __restrict__ sf,          // [B,256,200,176]
    const float* __restrict__ feat16,      // [N,16]
    const float* __restrict__ feat32,      // [N,32]
    const float* __restrict__ feat64a,     // [N,64]
    const float* __restrict__ feat64b,     // [N,64]
    const unsigned short* __restrict__ wbf,// [128,448] bf16
    const float* __restrict__ gamma,
    const float* __restrict__ beta,
    const float* __restrict__ rmean,
    const float* __restrict__ rvar,
    float* __restrict__ out)               // [N,128]
{
    extern __shared__ unsigned char smem_raw[];
    unsigned short* s_pf = (unsigned short*)smem_raw;                    // 128*456 u16
    unsigned short* s_w  = s_pf + TILE_M * ROW_LDS;                      // 128*456 u16
    float* s_scale = (float*)(s_w + C_OUT * ROW_LDS);                    // 128 f32
    float* s_shift = s_scale + C_OUT;                                    // 128 f32

    const int t    = threadIdx.x;
    const int wg   = blockIdx.x;
    const int pp   = t >> 1;               // 0..127 point within tile
    const int half = t & 1;
    const int p    = wg * TILE_M + pp;     // global point index

    // ---------------- Phase 1a: async-stage bf16 W into LDS -----------------
    // gfx1250 async global->LDS copy (tracked by ASYNCcnt); overlaps with the
    // bilinear gather below and uses no intermediate VGPR data registers.
    {
        unsigned lds_base;
        {
            __attribute__((address_space(3))) unsigned short* s_w3 =
                (__attribute__((address_space(3))) unsigned short*)s_w;
            lds_base = (unsigned)(unsigned long long)s_w3;
        }
        // 128 rows * 56 chunks-of-8-halfwords = 7168 chunks, 28 per thread
        #pragma unroll 4
        for (int it = 0; it < 28; ++it) {
            int chunk = t + it * 256;
            int row = chunk / 56;
            int col = chunk - row * 56;    // chunk-of-8 within row
            unsigned lds_addr = lds_base + (unsigned)(row * ROW_LDS + col * 8) * 2u;
            unsigned goff     = (unsigned)(row * K_PAD + col * 8) * 2u;
            asm volatile("global_load_async_to_lds_b128 %0, %1, %2"
                         :: "v"(lds_addr), "v"(goff), "s"(wbf)
                         : "memory");
        }
    }

    // ---------------- Phase 1b: BN scale/shift ------------------------------
    if (t < C_OUT) {
        float sc = gamma[t] * __frsqrt_rn(rvar[t] + BN_EPS);
        s_scale[t] = sc;
        s_shift[t] = beta[t] - rmean[t] * sc;
    }

    // ---------------- Phase 1c: point features (channels 0..175) ------------
    {
        unsigned short* dst_row = s_pf + pp * ROW_LDS;
        #pragma unroll 2
        for (int g0 = 0; g0 < 22; ++g0) {
            int g  = half * 22 + g0;       // 0..43 group of 4 channels
            int ch = g * 4;
            const float* src;
            if      (ch < 16)  src = feat16  + p * 16 + ch;
            else if (ch < 48)  src = feat32  + p * 32 + (ch - 16);
            else if (ch < 112) src = feat64a + p * 64 + (ch - 48);
            else               src = feat64b + p * 64 + (ch - 112);
            f32x4 v = *(const f32x4*)src;
            u32x2 st;
            st.x = (unsigned int)f32_to_bf16(v[0]) | ((unsigned int)f32_to_bf16(v[1]) << 16);
            st.y = (unsigned int)f32_to_bf16(v[2]) | ((unsigned int)f32_to_bf16(v[3]) << 16);
            *(u32x2*)(dst_row + ch) = st;
        }
        // zero K-pad (channels 432..447): 8 halfwords per helper thread
        u32x4 z; z.x = 0u; z.y = 0u; z.z = 0u; z.w = 0u;
        *(u32x4*)(dst_row + C_IN + half * 8) = z;
    }

    // ---------------- Phase 1d: bilinear BEV gather (channels 176..431) -----
    {
        float bx = coords[p * 4 + 0];
        float px = coords[p * 4 + 1];
        float py = coords[p * 4 + 2];
        int   b  = (int)bx;
        // x = (px - 0)/0.05/8 ; y = (py + 40)/0.05/8
        float x = px * 2.5f;
        float y = (py + 40.0f) * 2.5f;
        int ix = (int)floorf(x);
        int iy = (int)floorf(y);
        int x0 = min(max(ix,     0), W_G - 1);
        int x1 = min(max(ix + 1, 0), W_G - 1);
        int y0 = min(max(iy,     0), H_G - 1);
        int y1 = min(max(iy + 1, 0), H_G - 1);
        float xf0 = (float)x0, xf1 = (float)x1;
        float yf0 = (float)y0, yf1 = (float)y1;
        float wa = (xf1 - x) * (yf1 - y);
        float wb = (xf1 - x) * (y - yf0);
        float wc = (x - xf0) * (yf1 - y);
        float wd = (x - xf0) * (y - yf0);

        const float* base = sf + (size_t)b * (C_BEV * PLANE);
        int offA = y0 * W_G + x0;
        int offB = y1 * W_G + x0;
        int offC = y0 * W_G + x1;
        int offD = y1 * W_G + x1;

        unsigned short* dst_row = s_pf + pp * ROW_LDS + C_PT;
        int cbase = half * 128;
        #pragma unroll 4
        for (int i = 0; i < 128; i += 2) {
            int c = cbase + i;
            const float* p0 = base + (size_t)c * PLANE;
            const float* p1 = p0 + PLANE;
            float v0 = wa * p0[offA] + wb * p0[offB] + wc * p0[offC] + wd * p0[offD];
            float v1 = wa * p1[offA] + wb * p1[offB] + wc * p1[offC] + wd * p1[offD];
            unsigned int pk = (unsigned int)f32_to_bf16(v0) |
                              ((unsigned int)f32_to_bf16(v1) << 16);
            *(unsigned int*)(dst_row + c) = pk;
        }
    }

    // Wait for async W staging, then make tile visible to all waves.
    asm volatile("s_wait_asynccnt 0x0" ::: "memory");
    __syncthreads();

    // ---------------- Phase 2: WMMA GEMM, software pipelined ----------------
    // wave w -> points [w*16, w*16+16), all 8 n-tiles of 16 output channels.
    const int lane = t & 31;
    const int wave = t >> 5;
    const int m0   = wave * 16;
    const int lm   = lane & 15;
    const int lhi  = lane >> 4;

    f32x8 acc[8];
    #pragma unroll
    for (int nt = 0; nt < 8; ++nt)
        #pragma unroll
        for (int j = 0; j < 8; ++j) acc[nt][j] = 0.0f;

    const unsigned short* a_row  = s_pf + (m0 + lm) * ROW_LDS;
    const int kbA_off = lhi << 3;    // A: this half-wave holds K = b..b+7, b+16..b+23
    const int kbB_off = lhi << 3;    // B: this half-wave holds K = b..b+7, b+16..b+23

    BF16Frag af[2];                  // double-buffered A fragments
    BF16Frag bq[4];                  // 4-deep rotating B fragment queue

    auto loadA = [&](int kt, BF16Frag& f) {
        const unsigned short* pA = a_row + kt * 32 + kbA_off;
        f.u.lo = *(const u32x4*)pA;          // K = b .. b+7
        f.u.hi = *(const u32x4*)(pA + 16);   // K = b+16 .. b+23
    };
    auto loadB = [&](int g, BF16Frag& f) {   // g = kt*8 + nt
        int kt = g >> 3, nt = g & 7;
        const unsigned short* pB = s_w + (nt * 16 + lm) * ROW_LDS + kt * 32 + kbB_off;
        f.u.lo = *(const u32x4*)pB;          // K = b .. b+7
        f.u.hi = *(const u32x4*)(pB + 16);   // K = b+16 .. b+23
    };

    loadA(0, af[0]);
    loadB(0, bq[0]);
    loadB(1, bq[1]);
    loadB(2, bq[2]);

    #pragma unroll
    for (int kt = 0; kt < KT_N; ++kt) {
        if (kt + 1 < KT_N) loadA(kt + 1, af[(kt + 1) & 1]);   // prefetch next A
        #pragma unroll
        for (int nt = 0; nt < 8; ++nt) {
            const int g = kt * 8 + nt;
            if (g + 3 < KT_N * 8) loadB(g + 3, bq[(g + 3) & 3]);  // prefetch depth 3
            acc[nt] = __builtin_amdgcn_wmma_f32_16x16x32_bf16(
                /*neg_a=*/false, af[kt & 1].v, /*neg_b=*/false, bq[g & 3].v,
                /*c_mod=*/(short)0, acc[nt],
                /*reuse_a=*/false, /*reuse_b=*/false);
        }
    }

    // ---------------- Epilogue: BN + ReLU + store ---------------------------
    // D layout: VGPR j -> lanes 0-15: (M=j, N=lane); lanes 16-31: (M=j+8, N=lane-16)
    #pragma unroll
    for (int nt = 0; nt < 8; ++nt) {
        int o  = nt * 16 + lm;
        float sc = s_scale[o];
        float sh = s_shift[o];
        #pragma unroll
        for (int j = 0; j < 8; ++j) {
            int m = m0 + j + (lhi << 3);
            float v = acc[nt][j] * sc + sh;
            v = fmaxf(v, 0.0f);
            out[(size_t)(wg * TILE_M + m) * C_OUT + o] = v;
        }
    }
}

// ---------------------------------------------------------------------------
extern "C" void kernel_launch(void* const* d_in, const int* in_sizes, int n_in,
                              void* d_out, int out_size, void* d_ws, size_t ws_size,
                              hipStream_t stream) {
    const float* coords  = (const float*)d_in[0];
    const float* sf      = (const float*)d_in[1];
    const float* feat16  = (const float*)d_in[2];
    const float* feat32  = (const float*)d_in[3];
    const float* feat64a = (const float*)d_in[4];
    const float* feat64b = (const float*)d_in[5];
    const float* W       = (const float*)d_in[6];
    const float* gamma   = (const float*)d_in[7];
    const float* beta    = (const float*)d_in[8];
    const float* rmean   = (const float*)d_in[9];
    const float* rvar    = (const float*)d_in[10];
    float* out = (float*)d_out;

    unsigned short* wbf = (unsigned short*)d_ws;   // 128*448 bf16 = 114688 B

    // Raise dynamic LDS cap (CDNA5 WGP: up to 320KB per workgroup).
    const size_t smem = (size_t)(TILE_M * ROW_LDS + C_OUT * ROW_LDS) * 2 + 2 * C_OUT * 4;
    (void)hipFuncSetAttribute((const void*)fusion_kernel,
                              hipFuncAttributeMaxDynamicSharedMemorySize, (int)smem);

    wcvt_kernel<<<dim3((C_OUT * K_PAD) / 256), dim3(256), 0, stream>>>(W, wbf);
    fusion_kernel<<<dim3(N_PTS / TILE_M), dim3(256), smem, stream>>>(
        coords, sf, feat16, feat32, feat64a, feat64b, wbf,
        gamma, beta, rmean, rvar, out);
}
